// TemporalFusionTransformer_48473000902850
// MI455X (gfx1250) — matrix operand order repaired and checked by reading, not compiled, not run on hardware
//
#include <hip/hip_runtime.h>
#include <math.h>

typedef __attribute__((ext_vector_type(16))) _Float16 v16h;
typedef __attribute__((ext_vector_type(8)))  _Float16 v8h;
typedef __attribute__((ext_vector_type(8)))  float    v8f;

#define B_N 4096
#define T_N 256
#define D_N 8
#define H_N 64
#define DIN 9   // D + 1 (features + prev_delta)

// All weights staged once in LDS as f16 [N][Kpad]; activations as f16 [M][Kpad].
struct __align__(16) Smem {
    _Float16 w1  [64 * 32];    // W1  (64 x 9)  K padded to 32
    _Float16 w2  [16 * 64];    // W2  (9 x 64)  N padded to 16
    _Float16 wih0[256 * 32];   // Wih0 (256 x 9) K padded to 32
    _Float16 whh0[256 * 64];
    _Float16 wih1[256 * 64];
    _Float16 whh1[256 * 64];
    _Float16 xraw[16 * 32];    // [features, prev_delta, 0-pad]
    _Float16 xsel[16 * 32];    // after variable selection
    _Float16 relu[16 * 64];
    _Float16 h0  [16 * 64];
    _Float16 h1  [16 * 64];
    float    logits[16 * 16];
    float    b2s[16];
    float    wos[64];
    float    prevdelta[16];
};

// A fragment (16x32 f16). Per ISA layout: lane holds row M=lane&15; halves
// 0..7 -> K = kbase + 8*(lane>>4) + j ; halves 8..15 -> +16. Two contiguous
// 16B runs -> two ds_load_b128.
__device__ __forceinline__ v16h ldsA(const _Float16* base, int strideh, int kbase, int lane) {
    int m   = lane & 15;
    int khi = (lane >> 4) * 8;
    const _Float16* p = base + m * strideh + kbase + khi;
    v8h lo = *(const v8h*)p;
    v8h hi = *(const v8h*)(p + 16);
    v16h r;
#pragma unroll
    for (int i = 0; i < 8; ++i) { r[i] = lo[i]; r[i + 8] = hi[i]; }
    return r;
}

// B fragment (32x16 f16, K x N). Per ISA layout: lane holds column
// N = nbase + (lane&15), halves 0..7 -> K = kbase + 8*(lane>>4) + j ;
// halves 8..15 -> +16 (same K order as the A fragment).
__device__ __forceinline__ v16h ldsB(const _Float16* base, int strideh, int nbase, int kbase, int lane) {
    int n = nbase + (lane & 15);
    int k = kbase + ((lane >> 4) << 3);
    const _Float16* p = base + n * strideh + k;
    v8h lo = *(const v8h*)p;
    v8h hi = *(const v8h*)(p + 16);
    v16h r;
#pragma unroll
    for (int i = 0; i < 8; ++i) { r[i] = lo[i]; r[i + 8] = hi[i]; }
    return r;
}

__device__ __forceinline__ v8f wmma16(v16h a, v16h b, v8f c) {
    return __builtin_amdgcn_wmma_f32_16x16x32_f16(false, a, false, b, (short)0, c, false, false);
}

__device__ __forceinline__ float sigm(float x) { return 1.0f / (1.0f + __expf(-x)); }

__global__ void __launch_bounds__(128)
tft_recurrent_kernel(const float* __restrict__ features,
                     const float* __restrict__ W1,   const float* __restrict__ b1,
                     const float* __restrict__ W2,   const float* __restrict__ b2,
                     const float* __restrict__ Wih0, const float* __restrict__ Whh0,
                     const float* __restrict__ bih0, const float* __restrict__ bhh0,
                     const float* __restrict__ Wih1, const float* __restrict__ Whh1,
                     const float* __restrict__ bih1, const float* __restrict__ bhh1,
                     const float* __restrict__ Wo,   const float* __restrict__ bo,
                     float* __restrict__ out)
{
    extern __shared__ char smem_raw[];
    Smem* s = (Smem*)smem_raw;

    const int tid  = threadIdx.x;        // 128 threads = 4 wave32
    const int lane = tid & 31;
    const int wv   = tid >> 5;           // wave id 0..3 -> hidden cols 16*wv..
    const int b0   = blockIdx.x * 16;    // batch tile base

    // ---------------- one-time staging: weights -> LDS f16 ----------------
    for (int i = tid; i < 64 * 32; i += 128)  { int n = i >> 5, k = i & 31; s->w1[i]   = (k < DIN) ? (_Float16)W1[n * DIN + k]   : (_Float16)0.f; }
    for (int i = tid; i < 16 * 64; i += 128)  { int n = i >> 6, k = i & 63; s->w2[i]   = (n < DIN) ? (_Float16)W2[n * 64 + k]    : (_Float16)0.f; }
    for (int i = tid; i < 256 * 32; i += 128) { int n = i >> 5, k = i & 31; s->wih0[i] = (k < DIN) ? (_Float16)Wih0[n * DIN + k] : (_Float16)0.f; }
    for (int i = tid; i < 256 * 64; i += 128) s->whh0[i] = (_Float16)Whh0[i];
    for (int i = tid; i < 256 * 64; i += 128) s->wih1[i] = (_Float16)Wih1[i];
    for (int i = tid; i < 256 * 64; i += 128) s->whh1[i] = (_Float16)Whh1[i];
    for (int i = tid; i < 16 * 32; i += 128)  { s->xraw[i] = (_Float16)0.f; s->xsel[i] = (_Float16)0.f; }
    for (int i = tid; i < 16 * 64; i += 128)  { s->h0[i]   = (_Float16)0.f; s->h1[i]   = (_Float16)0.f; }
    if (tid < 16) { s->b2s[tid] = (tid < DIN) ? b2[tid] : 0.f; s->prevdelta[tid] = 0.f; }
    if (tid < 64) s->wos[tid] = Wo[tid];

    // per-lane constants: this lane owns gate/hidden column ncol of its wave tile
    const int   ncol = wv * 16 + (lane & 15);
    const float b1v  = b1[ncol];
    float bias0[4], bias1[4];
#pragma unroll
    for (int q = 0; q < 4; ++q) {
        bias0[q] = bih0[q * 64 + ncol] + bhh0[q * 64 + ncol];
        bias1[q] = bih1[q * 64 + ncol] + bhh1[q * 64 + ncol];
    }
    const float bov  = bo[0];
    const int   mrow = 8 * (lane >> 4);  // C-frag: VGPR r holds row M = r + mrow

    v8f c0 = {}; v8f c1 = {};            // cell states live in registers
    __syncthreads();

    for (int t = 0; t < T_N; ++t) {
        // -- 1: build x = [features_t, prev_delta] (f16, K-padded with zeros)
        {
            int m = tid >> 3, d = tid & 7;
            s->xraw[m * 32 + d] = (_Float16)features[((size_t)(b0 + m) * T_N + t) * D_N + d];
            if (tid < 16) s->xraw[tid * 32 + 8] = (_Float16)s->prevdelta[tid];
        }
        __syncthreads();

        // -- 2: var-selection layer 1: relu(x @ W1^T + b1)
        {
            v16h a = ldsA(s->xraw, 32, 0, lane);
            v8f acc = {};
            acc = wmma16(a, ldsB(s->w1, 32, wv * 16, 0, lane), acc);
#pragma unroll
            for (int r = 0; r < 8; ++r) {
                float v = acc[r] + b1v;
                s->relu[(r + mrow) * 64 + ncol] = (_Float16)(v > 0.f ? v : 0.f);
            }
        }
        __syncthreads();

        // -- 3: var-selection layer 2 logits: relu @ W2^T  (K = 64)
        {
            v16h a0 = ldsA(s->relu, 64, 0, lane);
            v16h a1 = ldsA(s->relu, 64, 32, lane);
            v8f acc = {};
            acc = wmma16(a0, ldsB(s->w2, 64, 0, 0, lane), acc);
            acc = wmma16(a1, ldsB(s->w2, 64, 0, 32, lane), acc);
            if (wv == 0) {
#pragma unroll
                for (int r = 0; r < 8; ++r) s->logits[(r + mrow) * 16 + (lane & 15)] = acc[r];
            }
        }
        __syncthreads();

        // -- 4: softmax over DIN + variable selection (one thread per row)
        if (tid < 16) {
            float v[DIN]; float mx = -1e30f;
#pragma unroll
            for (int n = 0; n < DIN; ++n) { v[n] = s->logits[tid * 16 + n] + s->b2s[n]; mx = fmaxf(mx, v[n]); }
            float ssum = 0.f;
#pragma unroll
            for (int n = 0; n < DIN; ++n) { v[n] = __expf(v[n] - mx); ssum += v[n]; }
            float inv = 1.f / ssum;
#pragma unroll
            for (int n = 0; n < DIN; ++n)
                s->xsel[tid * 32 + n] = (_Float16)((float)s->xraw[tid * 32 + n] * v[n] * inv);
        }
        __syncthreads();

        // -- 5: LSTM layer 0 (gates i,f,g,o = quarters of 256 output cols)
        {
            v16h aS  = ldsA(s->xsel, 32, 0, lane);
            v16h aH0 = ldsA(s->h0, 64, 0, lane);    // previous h0
            v16h aH1 = ldsA(s->h0, 64, 32, lane);
            v8f g[4];
#pragma unroll
            for (int q = 0; q < 4; ++q) {
                int nb = q * 64 + wv * 16;
                v8f acc = {};
                acc = wmma16(aS,  ldsB(s->wih0, 32, nb, 0, lane), acc);
                acc = wmma16(aH0, ldsB(s->whh0, 64, nb, 0, lane), acc);
                acc = wmma16(aH1, ldsB(s->whh0, 64, nb, 32, lane), acc);
                g[q] = acc;
            }
            __syncthreads();  // all waves finished reading old h0
#pragma unroll
            for (int r = 0; r < 8; ++r) {
                float iv = sigm(g[0][r] + bias0[0]);
                float fv = sigm(g[1][r] + bias0[1]);
                float gv = tanhf(g[2][r] + bias0[2]);
                float ov = sigm(g[3][r] + bias0[3]);
                float cc = fv * c0[r] + iv * gv;
                c0[r] = cc;
                s->h0[(r + mrow) * 64 + ncol] = (_Float16)(ov * tanhf(cc));
            }
        }
        __syncthreads();  // new h0 visible

        // -- 6: LSTM layer 1
        {
            v16h aX0 = ldsA(s->h0, 64, 0, lane);   // new h0
            v16h aX1 = ldsA(s->h0, 64, 32, lane);
            v16h aH0 = ldsA(s->h1, 64, 0, lane);   // previous h1
            v16h aH1 = ldsA(s->h1, 64, 32, lane);
            v8f g[4];
#pragma unroll
            for (int q = 0; q < 4; ++q) {
                int nb = q * 64 + wv * 16;
                v8f acc = {};
                acc = wmma16(aX0, ldsB(s->wih1, 64, nb, 0, lane), acc);
                acc = wmma16(aX1, ldsB(s->wih1, 64, nb, 32, lane), acc);
                acc = wmma16(aH0, ldsB(s->whh1, 64, nb, 0, lane), acc);
                acc = wmma16(aH1, ldsB(s->whh1, 64, nb, 32, lane), acc);
                g[q] = acc;
            }
            __syncthreads();  // all waves finished reading old h1
#pragma unroll
            for (int r = 0; r < 8; ++r) {
                float iv = sigm(g[0][r] + bias1[0]);
                float fv = sigm(g[1][r] + bias1[1]);
                float gv = tanhf(g[2][r] + bias1[2]);
                float ov = sigm(g[3][r] + bias1[3]);
                float cc = fv * c1[r] + iv * gv;
                c1[r] = cc;
                s->h1[(r + mrow) * 64 + ncol] = (_Float16)(ov * tanhf(cc));
            }
        }
        __syncthreads();  // new h1 visible

        // -- 7: output projection delta = h1 @ Wo^T + bo (one thread per row)
        if (tid < 16) {
            float acc = bov;
#pragma unroll
            for (int n = 0; n < 64; ++n) acc += (float)s->h1[tid * 64 + n] * s->wos[n];
            out[(size_t)(b0 + tid) * T_N + t] = acc;
            s->prevdelta[tid] = acc;
        }
        __syncthreads();  // prevdelta ready for next step
    }
}

extern "C" void kernel_launch(void* const* d_in, const int* in_sizes, int n_in,
                              void* d_out, int out_size, void* d_ws, size_t ws_size,
                              hipStream_t stream) {
    (void)in_sizes; (void)n_in; (void)d_ws; (void)ws_size; (void)out_size;
    const float* features = (const float*)d_in[0];
    const float* W1   = (const float*)d_in[1];
    const float* b1   = (const float*)d_in[2];
    const float* W2   = (const float*)d_in[3];
    const float* b2   = (const float*)d_in[4];
    const float* Wih0 = (const float*)d_in[5];
    const float* Whh0 = (const float*)d_in[6];
    const float* bih0 = (const float*)d_in[7];
    const float* bhh0 = (const float*)d_in[8];
    const float* Wih1 = (const float*)d_in[9];
    const float* Whh1 = (const float*)d_in[10];
    const float* bih1 = (const float*)d_in[11];
    const float* bhh1 = (const float*)d_in[12];
    const float* Wo   = (const float*)d_in[13];
    const float* bo   = (const float*)d_in[14];
    float* out = (float*)d_out;

    const size_t smem = sizeof(Smem);  // ~127 KB dynamic LDS (<< 320 KB/WGP)
    hipFuncSetAttribute((const void*)tft_recurrent_kernel,
                        hipFuncAttributeMaxDynamicSharedMemorySize, (int)smem);
    tft_recurrent_kernel<<<B_N / 16, 128, smem, stream>>>(
        features, W1, b1, W2, b2,
        Wih0, Whh0, bih0, bhh0,
        Wih1, Whh1, bih1, bhh1,
        Wo, bo, out);
}
